// VectorQuantizer_59296318488693
// MI455X (gfx1250) — matrix operand order repaired and checked by reading, not compiled, not run on hardware
//
#include <hip/hip_runtime.h>
#include <hip/hip_bf16.h>

typedef __attribute__((ext_vector_type(16))) _Float16 v16h;
typedef __attribute__((ext_vector_type(8)))  _Float16 v8h;
typedef __attribute__((ext_vector_type(8)))  float    v8f;

#define N_TOK    32768      // 32 * 32 * 32 rows
#define E_DIM    64
#define N_CODE   1024
#define HW       1024       // 32*32 spatial per batch
#define ZQ_ELEMS 2097152    // 32*64*32*32
#define EMB_SCALE 1024.0f   // lift tiny codebook values into f16 range

// ---------------------------------------------------------------------------
// K1: fused distance + argmin via v_wmma_f32_16x16x32_f16.
// grid 256 blocks x 256 threads (8 waves). Block: 128 rows; wave: 16 rows.
// Compare t_k = 1024*||e_k||^2 - 2*(z . 1024*e_k)  (monotone in true distance).
// g-loop unrolled x4 so the compare chain of tile g co-executes (TRANS/VALU
// dual issue) with wmma of tiles g+1.. instead of sitting behind hazard nops.
// ---------------------------------------------------------------------------
__global__ __launch_bounds__(256) void vq_argmin(
    const float* __restrict__ z,      // [32][64][1024] (B,C,H*W)
    const float* __restrict__ emb,    // [1024][64]
    int* __restrict__ idx_out)        // [32768]
{
  // stride 72 halves = 144B (16B multiple, kills the stride-64 bank pattern)
  __shared__ __align__(16) _Float16 zs[128 * 72];   // 18 KB
  __shared__ __align__(16) _Float16 ebs[256 * 72];  // 36 KB (codebook chunk)
  __shared__ float e2s[N_CODE];                     // 4 KB

  const int tid  = threadIdx.x;
  const int wave = tid >> 5;
  const int lane = tid & 31;
  const int ncol = lane & 15;   // column (code) within 16-wide tile; also row m for A
  const int hi   = lane >> 4;   // half-wave selector
  const int n0   = blockIdx.x * 128;
  const int b    = n0 >> 10;    // 128-row tile never crosses a batch boundary
  const float* zb = z + (size_t)b * (E_DIM * HW) + (n0 & (HW - 1));

  // Stage z tile (f32 -> f16); consecutive threads read consecutive hw (coalesced)
  for (int i = tid; i < 128 * 64; i += 256) {
    int r = i & 127, ch = i >> 7;
    zs[r * 72 + ch] = (_Float16)zb[(size_t)ch * HW + r];
  }
  // Scaled squared norms of all codes (f32, exact ranking term)
  for (int k = tid; k < N_CODE; k += 256) {
    const float4* e4 = (const float4*)(emb + k * 64);
    float s = 0.f;
#pragma unroll
    for (int j = 0; j < 16; ++j) {
      float4 v = e4[j];
      s += v.x * v.x + v.y * v.y + v.z * v.z + v.w * v.w;
    }
    e2s[k] = s * EMB_SCALE;
  }
  __syncthreads();

  // A fragments (this wave's 16 rows, K=0..31 and K=32..63), per ISA 16-bit A layout:
  // lane m=lane&15, halves 0..7 -> K = k0 + hi*8 + j ; halves 8..15 -> K = k0 + 16 + hi*8 + j
  const int r0 = wave * 16;
  const _Float16* zrow = &zs[(r0 + ncol) * 72];
  union { v16h v; v8h h[2]; } A0, A1;
  A0.h[0] = *(const v8h*)(zrow + hi * 8);
  A0.h[1] = *(const v8h*)(zrow + 16 + hi * 8);
  A1.h[0] = *(const v8h*)(zrow + 32 + hi * 8);
  A1.h[1] = *(const v8h*)(zrow + 48 + hi * 8);

  float best[8];
  int   bidx[8];
#pragma unroll
  for (int i = 0; i < 8; ++i) { best[i] = 3.0e38f; bidx[i] = 0; }

  for (int chunk = 0; chunk < 4; ++chunk) {
    __syncthreads();  // previous chunk fully consumed
    const float* ech = emb + (size_t)chunk * 256 * 64;
    for (int i = tid; i < 256 * 64; i += 256) {
      int row = i >> 6, ch = i & 63;
      ebs[row * 72 + ch] = (_Float16)(ech[row * 64 + ch] * EMB_SCALE);
    }
    __syncthreads();

#pragma unroll 4
    for (int g = 0; g < 16; ++g) {
      const int cb16 = chunk * 256 + g * 16;
      // B layout (32x16 f16): same K order as A: halves 0..7 -> K = hi*8 + j ; halves 8..15 -> K = 16 + hi*8 + j
      const _Float16* erow = &ebs[(g * 16 + ncol) * 72];
      union { v16h v; v8h h[2]; } B0, B1;
      B0.h[0] = *(const v8h*)(erow + hi * 8);
      B0.h[1] = *(const v8h*)(erow + 16 + hi * 8);
      B1.h[0] = *(const v8h*)(erow + 32 + hi * 8);
      B1.h[1] = *(const v8h*)(erow + 48 + hi * 8);

      v8f acc = {};
      acc = __builtin_amdgcn_wmma_f32_16x16x32_f16(false, A0.v, false, B0.v,
                                                   (short)0, acc, false, false);
      acc = __builtin_amdgcn_wmma_f32_16x16x32_f16(false, A1.v, false, B1.v,
                                                   (short)0, acc, false, false);
      const float e2   = e2s[cb16 + ncol];
      const int   code = cb16 + ncol;
#pragma unroll
      for (int i = 0; i < 8; ++i) {           // C VGPR i -> row M = i + hi*8, col = ncol
        float t = e2 - 2.0f * acc[i];
        if (t < best[i]) { best[i] = t; bidx[i] = code; }  // strict < keeps first min
      }
    }
  }

  // Lexicographic (val, idx) min across the 16 lanes of each half-wave
#pragma unroll
  for (int off = 8; off >= 1; off >>= 1) {
#pragma unroll
    for (int i = 0; i < 8; ++i) {
      float ov = __shfl_xor(best[i], off, 32);
      int   oi = __shfl_xor(bidx[i], off, 32);
      if (ov < best[i] || (ov == best[i] && oi < bidx[i])) { best[i] = ov; bidx[i] = oi; }
    }
  }
  if (ncol == 0) {
#pragma unroll
    for (int i = 0; i < 8; ++i)
      idx_out[n0 + r0 + hi * 8 + i] = bidx[i];
  }
}

// ---------------------------------------------------------------------------
// K2a: z_q gather/scatter (exact f32 codebook) + per-block loss partial sums.
// 1024 blocks x 256 threads; block handles 32 consecutive rows (same batch b).
// ---------------------------------------------------------------------------
__global__ __launch_bounds__(256) void vq_zq_loss(
    const float* __restrict__ z, const float* __restrict__ emb,
    const int* __restrict__ idx, float* __restrict__ zq_out,
    float* __restrict__ lpart)
{
  __shared__ float red[256];
  const int t      = threadIdx.x;
  const int nb     = blockIdx.x;
  const int n_base = nb * 32;
  const int b      = n_base >> 10;
  const int hw0    = n_base & (HW - 1);
  const int hwo    = t & 31;
  const int idxn   = idx[n_base + hwo];
  float lsum = 0.f;
#pragma unroll
  for (int cc = 0; cc < 64; cc += 8) {
    int c = cc + (t >> 5);
    float  ev  = emb[idxn * 64 + c];
    size_t off = (size_t)b * 65536 + (size_t)c * HW + hw0 + hwo;  // coalesced over hwo
    float  zv  = z[off];
    zq_out[off] = ev;                 // z_q_st == z_q numerically (STE)
    float d = ev - zv;
    lsum += d * d;
  }
  red[t] = lsum;
  __syncthreads();
  for (int s = 128; s > 0; s >>= 1) {
    if (t < s) red[t] += red[t + s];
    __syncthreads();
  }
  if (t == 0) lpart[nb] = red[0];
}

// ---------------------------------------------------------------------------
// K2b: one-hot (dominant 128 MB stream) + indices as float.
// Region is only 8B-aligned (float offset 2097154) -> float2 stores.
// ---------------------------------------------------------------------------
__global__ __launch_bounds__(256) void vq_onehot(
    const int* __restrict__ idx, float* __restrict__ onehot,
    float* __restrict__ idxf_out)
{
  const int n    = blockIdx.x;
  const int t    = threadIdx.x;
  const int idxn = idx[n];
  const int base = t * 4;
  float2 va = make_float2(0.f, 0.f);
  float2 vb = make_float2(0.f, 0.f);
  const int d = idxn - base;
  if (d == 0)      va.x = 1.f;
  else if (d == 1) va.y = 1.f;
  else if (d == 2) vb.x = 1.f;
  else if (d == 3) vb.y = 1.f;
  float2* o = (float2*)(onehot + (size_t)n * N_CODE);
  o[2 * t]     = va;
  o[2 * t + 1] = vb;
  if (t == 0) idxf_out[n] = (float)idxn;
}

// ---------------------------------------------------------------------------
// K3: perplexity (LDS int histogram, order-independent) + final loss.
// ---------------------------------------------------------------------------
__global__ __launch_bounds__(1024) void vq_stats(
    const int* __restrict__ idx, const float* __restrict__ lpart,
    float* __restrict__ loss_out, float* __restrict__ perp_out)
{
  __shared__ int   hist[N_CODE];
  __shared__ float red[1024];
  const int t = threadIdx.x;
  hist[t] = 0;
  __syncthreads();
  for (int i = t; i < N_TOK; i += 1024) atomicAdd(&hist[idx[i]], 1);
  __syncthreads();
  float em = (float)hist[t] * (1.0f / (float)N_TOK);
  red[t] = em * logf(em + 1e-10f);
  __syncthreads();
  for (int s = 512; s > 0; s >>= 1) {
    if (t < s) red[t] += red[t + s];
    __syncthreads();
  }
  if (t == 0) perp_out[0] = expf(-red[0]);
  __syncthreads();
  red[t] = lpart[t];
  __syncthreads();
  for (int s = 512; s > 0; s >>= 1) {
    if (t < s) red[t] += red[t + s];
    __syncthreads();
  }
  if (t == 0) loss_out[0] = 1.25f * red[0] / (float)ZQ_ELEMS;  // (1+beta)*mean
}

// ---------------------------------------------------------------------------
extern "C" void kernel_launch(void* const* d_in, const int* in_sizes, int n_in,
                              void* d_out, int out_size, void* d_ws, size_t ws_size,
                              hipStream_t stream) {
  (void)in_sizes; (void)n_in; (void)out_size; (void)ws_size;
  const float* z   = (const float*)d_in[0];   // (32,64,32,32) f32
  const float* emb = (const float*)d_in[1];   // (1024,64) f32

  float* out      = (float*)d_out;
  float* loss_out = out;                                   // [1]
  float* zq_out   = out + 1;                               // [2097152]
  float* perp_out = out + 1 + ZQ_ELEMS;                    // [1]
  float* onehot   = out + 2 + ZQ_ELEMS;                    // [32768*1024]
  float* idxf_out = onehot + (size_t)N_TOK * N_CODE;       // [32768]

  int*   idx   = (int*)d_ws;                               // 32768 ints
  float* lpart = (float*)((char*)d_ws + N_TOK * sizeof(int)); // 1024 floats

  vq_argmin <<<dim3(N_TOK / 128), dim3(256), 0, stream>>>(z, emb, idx);
  vq_zq_loss<<<dim3(N_TOK / 32),  dim3(256), 0, stream>>>(z, emb, idx, zq_out, lpart);
  vq_onehot <<<dim3(N_TOK),       dim3(256), 0, stream>>>(idx, onehot, idxf_out);
  vq_stats  <<<dim3(1),           dim3(1024), 0, stream>>>(idx, lpart, loss_out, perp_out);
}
